// acLSTM_51213190038154
// MI455X (gfx1250) — matrix operand order repaired and checked by reading, not compiled, not run on hardware
//
#include <hip/hip_runtime.h>
#include <math.h>

typedef unsigned short u16;
typedef __attribute__((ext_vector_type(16))) __bf16 v16bf;
typedef __attribute__((ext_vector_type(8)))  float  v8f;
typedef unsigned int u32x4 __attribute__((ext_vector_type(4)));

union Frag { v16bf v; u32x4 q[2]; };

#define HID    1024
#define BSZ    64
#define INF_   227
#define INPAD  256
#define OUTPAD 240
#define TSEQ   200
#define NBLK   64          // persistent grid size (one block per 16 hidden cols)

__device__ __forceinline__ u16 f2bf(float f) {
    unsigned int u = __float_as_uint(f);
    unsigned int r = u + 0x7FFFu + ((u >> 16) & 1u);   // round-to-nearest-even
    return (u16)(r >> 16);
}

// ---------------------------------------------------------------------------
// Pack fp32 weight matrix W[N][K] into WMMA bf16 B-fragment-major layout.
// Fragment f = n_tile*nKt + kt is a 32(K)x16(N) block: lane L owns column
// N = n_tile*16 + (L&15), K = kt*32 + (L>=16?8:0) + (i<8 ? i : i+8), 16 contiguous bf16
// per lane (32B -> two global_load_b128 in the consumer).
// ---------------------------------------------------------------------------
__global__ __launch_bounds__(256) void pack_weights(
    const float* __restrict__ W, u16* __restrict__ pack,
    int N, int K, int Kpad, int nFrag)
{
    int w = threadIdx.x >> 5, lane = threadIdx.x & 31;
    int f = blockIdx.x * 8 + w;
    if (f >= nFrag) return;
    int nKt = Kpad >> 5;
    int kt = f % nKt, nt = f / nKt;
    int n  = nt * 16 + (lane & 15);
    int kb = kt * 32 + ((lane >> 4) << 3);
    u16* dst = pack + (size_t)f * 512 + lane * 16;
    const float* src = W + (size_t)n * K;
    #pragma unroll
    for (int i = 0; i < 16; i++) {
        int k = kb + (i < 8 ? i : i + 8);
        dst[i] = (n < N && k < K) ? f2bf(src[k]) : (u16)0;
    }
}

// Convert ALL ground-truth frames up front: gt index fi <-> t=(fi/5)*10+fi%5
__global__ __launch_bounds__(256) void frames_conv_all(
    const float* __restrict__ real_seq, u16* __restrict__ gt_frames)
{
    int idx = blockIdx.x * 256 + threadIdx.x;       // 100*64*256 total
    if (idx >= 100 * BSZ * INPAD) return;
    int fi  = idx / (BSZ * INPAD);
    int rem = idx % (BSZ * INPAD);
    int b = rem >> 8, k = rem & 255;
    int t = (fi / 5) * 10 + (fi % 5);
    float v = (k < INF_) ? real_seq[((size_t)b * TSEQ + t) * INF_ + k] : 0.f;
    gt_frames[idx] = f2bf(v);
}

// ---------------------------------------------------------------------------
// Device-wide barrier: monotone counter, release-add + acquire-spin (s_sleep)
// ---------------------------------------------------------------------------
__device__ __forceinline__ void grid_barrier(unsigned int* bar, unsigned int target)
{
    __threadfence();                 // make this thread's stores visible
    __syncthreads();
    if (threadIdx.x == 0) {
        __hip_atomic_fetch_add(bar, 1u, __ATOMIC_RELEASE, __HIP_MEMORY_SCOPE_AGENT);
        unsigned int v;
        do {
            __builtin_amdgcn_s_sleep(1);
            v = __hip_atomic_load(bar, __ATOMIC_ACQUIRE, __HIP_MEMORY_SCOPE_AGENT);
        } while (v < target);
    }
    __syncthreads();
    __threadfence();                 // invalidate stale near caches for all threads
}

// ---------------------------------------------------------------------------
// K-range slice of gates += A[64 x K-slice] * B-pack. Each wave owns one gate
// and one K-quarter; 4 accumulators (batch tiles) reuse every B fragment 4x.
// ---------------------------------------------------------------------------
__device__ __forceinline__ void gemm_part(
    v8f acc[4], const u16* __restrict__ src, int K,
    const u16* __restrict__ pack, int lane, int kt0, int kt1)
{
    const int lh = lane >> 4, ln = lane & 15;
    const u16* abase = src + (size_t)ln * K + lh * 8;
    for (int kt = kt0; kt < kt1; ++kt) {
        Frag b;
        const u32x4* bp = (const u32x4*)(pack + (size_t)kt * 512 + lane * 16);
        b.q[0] = bp[0]; b.q[1] = bp[1];
        if (kt + 1 < kt1)
            __builtin_prefetch(pack + (size_t)(kt + 1) * 512 + lane * 16, 0, 3);
        #pragma unroll
        for (int mt = 0; mt < 4; ++mt) {
            Frag a;
            const u32x4* ap = (const u32x4*)(abase + (size_t)mt * 16 * K + kt * 32);
            a.q[0] = ap[0];   // K = kt*32 + (lh?8:0) + 0..7
            a.q[1] = ap[2];   // K = ... + 16..23
            acc[mt] = __builtin_amdgcn_wmma_f32_16x16x32_bf16(
                false, a.v, false, b.v, (short)0, acc[mt], false, false);
        }
    }
}

// One LSTM layer for this block's 16 hidden columns. 16 waves = 4 gates x 4
// K-quarters; partials reduced in LDS, then fused sigmoid/tanh + c/h update.
__device__ __forceinline__ void cell_phase(
    float* gates,                        // shared [4 parts * 4 gates][64*16]
    const u16* __restrict__ x, int Kx,
    const u16* __restrict__ pWih,
    const u16* __restrict__ h_in,
    const u16* __restrict__ pWhh,
    const float* __restrict__ bias,
    float* __restrict__ c_state,
    u16* __restrict__ h_out)
{
    const int tid = threadIdx.x;
    const int wv = tid >> 5, lane = tid & 31;
    const int g = wv & 3, part = wv >> 2;
    const int lh = lane >> 4, ln = lane & 15;
    const int j0 = blockIdx.x * 16;

    v8f acc[4] = {};
    const int nKtx = Kx >> 5, nKth = HID >> 5;
    gemm_part(acc, x, Kx,
              pWih + (size_t)(g * 64 + blockIdx.x) * nKtx * 512, lane,
              part * (nKtx >> 2), (part + 1) * (nKtx >> 2));
    gemm_part(acc, h_in, HID,
              pWhh + (size_t)(g * 64 + blockIdx.x) * nKth * 512, lane,
              part * (nKth >> 2), (part + 1) * (nKth >> 2));

    #pragma unroll
    for (int mt = 0; mt < 4; ++mt)
        #pragma unroll
        for (int r = 0; r < 8; ++r)
            gates[(part * 4 + g) * 1024 + (mt * 16 + lh * 8 + r) * 16 + ln] = acc[mt][r];
    __syncthreads();

    for (int idx = tid; idx < BSZ * 16; idx += 512) {
        int j = idx & 15;
        float gi = gates[ 0 * 1024 + idx] + gates[ 4 * 1024 + idx]
                 + gates[ 8 * 1024 + idx] + gates[12 * 1024 + idx] + bias[          j0 + j];
        float gf = gates[ 1 * 1024 + idx] + gates[ 5 * 1024 + idx]
                 + gates[ 9 * 1024 + idx] + gates[13 * 1024 + idx] + bias[    HID + j0 + j];
        float gg = gates[ 2 * 1024 + idx] + gates[ 6 * 1024 + idx]
                 + gates[10 * 1024 + idx] + gates[14 * 1024 + idx] + bias[2 * HID + j0 + j];
        float go = gates[ 3 * 1024 + idx] + gates[ 7 * 1024 + idx]
                 + gates[11 * 1024 + idx] + gates[15 * 1024 + idx] + bias[3 * HID + j0 + j];
        float i_ = 1.f / (1.f + __expf(-gi));
        float f_ = 1.f / (1.f + __expf(-gf));
        float o_ = 1.f / (1.f + __expf(-go));
        float g_ = tanhf(gg);
        size_t ci = (size_t)(idx >> 4) * HID + j0 + j;
        float cn = f_ * c_state[ci] + i_ * g_;
        c_state[ci] = cn;
        h_out[ci] = f2bf(o_ * tanhf(cn));
    }
}

// Decoder: out = h2 @ W_dec^T + b_dec (64 x 227, K=1024). All 16 waves used:
// wave = (batch tile, K quarter); partials reduced in LDS. Blocks 0..14 only.
__device__ __forceinline__ void decoder_phase(
    float* gates, const u16* __restrict__ h2, const u16* __restrict__ packW,
    const float* __restrict__ bdec, float* __restrict__ outp,
    u16* __restrict__ prev_out)
{
    const int tid = threadIdx.x;
    const int wv = tid >> 5, lane = tid & 31;
    const int mt = wv & 3, part = wv >> 2;
    const int lh = lane >> 4, ln = lane & 15;
    const int n_tile = blockIdx.x;
    v8f acc = {};
    const u16* packB = packW + (size_t)n_tile * 32 * 512 + lane * 16;
    const u16* abase = h2 + (size_t)(mt * 16 + ln) * HID + lh * 8;
    for (int kt = part * 8; kt < (part + 1) * 8; ++kt) {
        Frag b;
        const u32x4* bp = (const u32x4*)(packB + (size_t)kt * 512);
        b.q[0] = bp[0]; b.q[1] = bp[1];
        Frag a;
        const u32x4* ap = (const u32x4*)(abase + kt * 32);
        a.q[0] = ap[0]; a.q[1] = ap[2];
        acc = __builtin_amdgcn_wmma_f32_16x16x32_bf16(
            false, a.v, false, b.v, (short)0, acc, false, false);
    }
    #pragma unroll
    for (int r = 0; r < 8; ++r)
        gates[(part * 4 + mt) * 256 + (lh * 8 + r) * 16 + ln] = acc[r];
    __syncthreads();

    for (int idx = tid; idx < 1024; idx += 512) {
        int m = idx >> 8, within = idx & 255;
        float v = gates[(0 * 4 + m) * 256 + within] + gates[(1 * 4 + m) * 256 + within]
                + gates[(2 * 4 + m) * 256 + within] + gates[(3 * 4 + m) * 256 + within];
        int N  = n_tile * 16 + (within & 15);
        int bb = m * 16 + (within >> 4);
        if (N < INF_) {
            v += bdec[N];
            outp[(size_t)bb * (TSEQ * INF_) + N] = v;
        }
        prev_out[bb * INPAD + N] = f2bf(v);
    }
}

// ---------------------------------------------------------------------------
struct Params {
    const u16 *gt, *pWih1, *pWhh1, *pWih2, *pWhh2, *pWih3, *pWhh3, *pWdec;
    const float *b1, *b2, *b3, *bdec;
    u16 *h0[2], *h1[2], *h2[2];
    float *c0, *c1, *c2;
    u16 *prev;
    float *out;
    unsigned int *bar;
};

// Single persistent kernel: all 200 timesteps. 64 blocks x 512 threads is far
// below WGP count -> co-resident, spin barrier cannot deadlock. The post-
// decoder barrier is elided when step t+1 is teacher-forced (prev_out unused),
// overlapping the decoder of t with layer 1 of t+1.
__global__ __launch_bounds__(512) void lstm_persistent(Params P)
{
    __shared__ float gates[16 * 1024];   // 64KB: [Kpart][gate][64 batch][16 col]
    unsigned int phase = 0;
    #pragma unroll 1
    for (int t = 0; t < TSEQ; ++t) {
        const u16* x0 = ((t % 10) < 5)
            ? P.gt + (size_t)((t / 10) * 5 + (t % 5)) * (BSZ * INPAD)
            : P.prev;
        int pi = t & 1, po = pi ^ 1;
        cell_phase(gates, x0,       INPAD, P.pWih1, P.h0[pi], P.pWhh1, P.b1, P.c0, P.h0[po]);
        grid_barrier(P.bar, NBLK * ++phase);
        cell_phase(gates, P.h0[po], HID,   P.pWih2, P.h1[pi], P.pWhh2, P.b2, P.c1, P.h1[po]);
        grid_barrier(P.bar, NBLK * ++phase);
        cell_phase(gates, P.h1[po], HID,   P.pWih3, P.h2[pi], P.pWhh3, P.b3, P.c2, P.h2[po]);
        grid_barrier(P.bar, NBLK * ++phase);
        if (blockIdx.x < OUTPAD / 16)
            decoder_phase(gates, P.h2[po], P.pWdec, P.bdec,
                          P.out + (size_t)t * INF_, P.prev);
        // Barrier only if the NEXT step actually consumes prev_out.
        bool next_is_cond = (t + 1 < TSEQ) && (((t + 1) % 10) >= 5);
        if (next_is_cond)
            grid_barrier(P.bar, NBLK * ++phase);
    }
}

// ---------------------------------------------------------------------------
extern "C" void kernel_launch(void* const* d_in, const int* in_sizes, int n_in,
                              void* d_out, int out_size, void* d_ws, size_t ws_size,
                              hipStream_t stream)
{
    (void)in_sizes; (void)n_in; (void)out_size; (void)ws_size;
    const float* real_seq = (const float*)d_in[0];
    const float* W_ih1 = (const float*)d_in[1];
    const float* W_hh1 = (const float*)d_in[2];
    const float* b1    = (const float*)d_in[3];
    const float* W_ih2 = (const float*)d_in[4];
    const float* W_hh2 = (const float*)d_in[5];
    const float* b2    = (const float*)d_in[6];
    const float* W_ih3 = (const float*)d_in[7];
    const float* W_hh3 = (const float*)d_in[8];
    const float* b3    = (const float*)d_in[9];
    const float* W_dec = (const float*)d_in[10];
    const float* b_dec = (const float*)d_in[11];
    char* ws = (char*)d_ws;

    size_t off = 0;
    auto sub = [&](size_t bytes) -> void* {
        off = (off + 255) & ~(size_t)255;
        void* p = (void*)(ws + off);
        off += bytes;
        return p;
    };

    // Packed bf16 weight fragments (~42.5MB: fully resident in 192MB L2)
    const int FR_IH1 = (INPAD / 32) * (4 * HID / 16);   // 2048
    const int FR_H   = (HID   / 32) * (4 * HID / 16);   // 8192
    const int FR_DEC = (HID   / 32) * (OUTPAD / 16);    // 480
    u16* pWih1 = (u16*)sub((size_t)FR_IH1 * 512 * 2);
    u16* pWhh1 = (u16*)sub((size_t)FR_H   * 512 * 2);
    u16* pWih2 = (u16*)sub((size_t)FR_H   * 512 * 2);
    u16* pWhh2 = (u16*)sub((size_t)FR_H   * 512 * 2);
    u16* pWih3 = (u16*)sub((size_t)FR_H   * 512 * 2);
    u16* pWhh3 = (u16*)sub((size_t)FR_H   * 512 * 2);
    u16* pWdec = (u16*)sub((size_t)FR_DEC * 512 * 2);
    u16* gtf   = (u16*)sub((size_t)100 * BSZ * INPAD * 2);   // pre-converted GT frames

    off = (off + 255) & ~(size_t)255;
    size_t state_begin = off;
    Params P;
    P.h0[0] = (u16*)sub(BSZ * HID * 2); P.h0[1] = (u16*)sub(BSZ * HID * 2);
    P.h1[0] = (u16*)sub(BSZ * HID * 2); P.h1[1] = (u16*)sub(BSZ * HID * 2);
    P.h2[0] = (u16*)sub(BSZ * HID * 2); P.h2[1] = (u16*)sub(BSZ * HID * 2);
    P.c0 = (float*)sub(BSZ * HID * 4);
    P.c1 = (float*)sub(BSZ * HID * 4);
    P.c2 = (float*)sub(BSZ * HID * 4);
    P.prev = (u16*)sub(BSZ * INPAD * 2);
    P.bar  = (unsigned int*)sub(256);
    size_t state_end = off;

    hipMemsetAsync(ws + state_begin, 0, state_end - state_begin, stream);

    pack_weights<<<dim3((FR_IH1 + 7) / 8), 256, 0, stream>>>(W_ih1, pWih1, 4 * HID, INF_, INPAD, FR_IH1);
    pack_weights<<<dim3((FR_H   + 7) / 8), 256, 0, stream>>>(W_hh1, pWhh1, 4 * HID, HID,  HID,  FR_H);
    pack_weights<<<dim3((FR_H   + 7) / 8), 256, 0, stream>>>(W_ih2, pWih2, 4 * HID, HID,  HID,  FR_H);
    pack_weights<<<dim3((FR_H   + 7) / 8), 256, 0, stream>>>(W_hh2, pWhh2, 4 * HID, HID,  HID,  FR_H);
    pack_weights<<<dim3((FR_H   + 7) / 8), 256, 0, stream>>>(W_ih3, pWih3, 4 * HID, HID,  HID,  FR_H);
    pack_weights<<<dim3((FR_H   + 7) / 8), 256, 0, stream>>>(W_hh3, pWhh3, 4 * HID, HID,  HID,  FR_H);
    pack_weights<<<dim3((FR_DEC + 7) / 8), 256, 0, stream>>>(W_dec, pWdec, INF_,    HID,  HID,  FR_DEC);
    frames_conv_all<<<dim3((100 * BSZ * INPAD + 255) / 256), 256, 0, stream>>>(real_seq, gtf);

    P.gt = gtf;
    P.pWih1 = pWih1; P.pWhh1 = pWhh1;
    P.pWih2 = pWih2; P.pWhh2 = pWhh2;
    P.pWih3 = pWih3; P.pWhh3 = pWhh3;
    P.pWdec = pWdec;
    P.b1 = b1; P.b2 = b2; P.b3 = b3; P.bdec = b_dec;
    P.out = (float*)d_out;

    lstm_persistent<<<dim3(NBLK), 512, 0, stream>>>(P);
}
